// WanAttentionBlock_4638564679738
// MI455X (gfx1250) — compile-verified, matrix operand order checked by reading, not run on hardware
//
#include <hip/hip_runtime.h>
#include <cstdint>
#include <cstddef>

#define S_LEN   4096
#define L_LEN   512
#define DMODEL  1536
#define NHEAD   12
#define HDIM    128
#define FFDIM   8960
#define RMS_EPS 1e-6f
#define ATTN_SCALE 0.08838834764831845f   // 128^-0.5

typedef __bf16 bf16;
typedef __attribute__((ext_vector_type(8)))  __bf16 bf16x8;
typedef __attribute__((ext_vector_type(16))) __bf16 bf16x16;
typedef __attribute__((ext_vector_type(8)))  float  f32x8;

// ---------------- helpers ----------------

__device__ __forceinline__ bf16 f2bf(float f) {
  unsigned u = __builtin_bit_cast(unsigned, f);
  unsigned r = u + 0x7FFFu + ((u >> 16) & 1u);       // round-to-nearest-even
  unsigned short s = (unsigned short)(r >> 16);
  return __builtin_bit_cast(bf16, s);
}

// Load a 16x32 bf16 A/B fragment from a row-major LDS tile.
// ISA layout (16-bit A-matrix 16x32): lanes 0-15 -> K 0-7 (v0-3) & 16-23 (v4-7),
// lanes 16-31 -> K 8-15 & 24-31. Two contiguous 16B chunks per lane.
__device__ __forceinline__ bf16x16 load_frag(const bf16* tile, int row0, int kOff, int stride) {
  int lane = threadIdx.x & 31;
  const bf16* p = tile + (size_t)(row0 + (lane & 15)) * stride + kOff + ((lane >> 4) << 3);
  bf16x8 lo = *(const bf16x8*)(p);
  bf16x8 hi = *(const bf16x8*)(p + 16);
  return __builtin_shufflevector(lo, hi, 0,1,2,3,4,5,6,7,8,9,10,11,12,13,14,15);
}

__device__ __forceinline__ f32x8 wmma_bf16(bf16x16 a, bf16x16 b, f32x8 c) {
  return __builtin_amdgcn_wmma_f32_16x16x32_bf16(false, a, false, b, (short)0, c, false, false);
}

__device__ __forceinline__ float sigmoid_fast(float u) {   // branchless, overflow-safe
  float eu = __expf(-fabsf(u));
  float s  = 1.0f / (1.0f + eu);        // value for u >= 0
  return (u >= 0.0f) ? s : (1.0f - s);
}

// CDNA5 async global->LDS copy, 16B per lane (GLOBAL_LOAD_ASYNC_TO_LDS_B128, ASYNCcnt).
// ldsOff: LDS byte offset (low 32 bits of generic pointer); gOff: byte offset from base.
__device__ __forceinline__ void async_b128(unsigned ldsOff, unsigned gOff, const void* base) {
  asm volatile("global_load_async_to_lds_b128 %0, %1, %2"
               :: "v"(ldsOff), "v"(gOff), "s"(base) : "memory");
}
__device__ __forceinline__ void wait_async0() {
  asm volatile("s_wait_asynccnt 0x0" ::: "memory");
}

// ---------------- weight pre-transpose + f32 -> bf16:  W[K][N] -> WT[N][K] ----------------
__global__ __launch_bounds__(256) void transpose_conv_kernel(
    const float* __restrict__ W, bf16* __restrict__ WT, int K, int N)
{
  __shared__ bf16 tile[64][72];
  int bk = blockIdx.y * 64, bn = blockIdx.x * 64;
  int tid = threadIdx.x;
#pragma unroll
  for (int i = 0; i < 4; i++) {                 // read 64x64 f32, coalesced along n
    int c = tid + i * 256;                      // [0,1024)
    int k = c >> 4, n0 = (c & 15) * 4;
    float4 f = *(const float4*)(W + (size_t)(bk + k) * N + bn + n0);
    tile[n0 + 0][k] = f2bf(f.x);
    tile[n0 + 1][k] = f2bf(f.y);
    tile[n0 + 2][k] = f2bf(f.z);
    tile[n0 + 3][k] = f2bf(f.w);
  }
  __syncthreads();
#pragma unroll
  for (int i = 0; i < 2; i++) {                 // write 64x64 bf16, coalesced along k
    int c = tid + i * 256;                      // [0,512)
    int n = c >> 3, k0 = (c & 7) * 8;
    *(uint4*)(WT + (size_t)(bn + n) * K + bk + k0) = *(const uint4*)(&tile[n][k0]);
  }
}

// ---------------- RMS (+ optional shift/scale modulation) -> bf16 ----------------
__global__ __launch_bounds__(256) void rmsmod_kernel(
    const float* __restrict__ in, const float* __restrict__ w,
    const float* __restrict__ shift, const float* __restrict__ scale,
    bf16* __restrict__ out)
{
  int row = blockIdx.x;
  int tid = threadIdx.x;
  const float* r = in + (size_t)row * DMODEL;
  float v[6]; float ss = 0.f;
#pragma unroll
  for (int j = 0; j < 6; j++) { v[j] = r[tid + j * 256]; ss += v[j] * v[j]; }
#pragma unroll
  for (int off = 16; off; off >>= 1) ss += __shfl_xor(ss, off, 32);
  __shared__ float red[8];
  if ((tid & 31) == 0) red[tid >> 5] = ss;
  __syncthreads();
  float tot = 0.f;
#pragma unroll
  for (int i = 0; i < 8; i++) tot += red[i];
  float rr = rsqrtf(tot * (1.0f / DMODEL) + RMS_EPS);
  bf16* o = out + (size_t)row * DMODEL;
#pragma unroll
  for (int j = 0; j < 6; j++) {
    int d = tid + j * 256;
    float val = v[j] * rr * w[d];
    if (scale) val = val * (1.0f + scale[d]) + shift[d];
    o[d] = f2bf(val);
  }
}

// ---------------- optional RMS + optional RoPE -> head-major bf16 ----------------
// tr==0: out[H][M][HD]   (Q / K layout)
// tr==1: out[H][HD][M]   (V transposed layout for attention B-fragments)
__global__ __launch_bounds__(256) void headprep_kernel(
    const float* __restrict__ in, const float* __restrict__ w,
    const float* __restrict__ freqs, bf16* __restrict__ out, int M, int tr)
{
  int row = blockIdx.x;
  int tid = threadIdx.x;
  const float* r = in + (size_t)row * DMODEL;
  __shared__ float vbuf[DMODEL];
  __shared__ float red[8];
  float v[6]; float ss = 0.f;
#pragma unroll
  for (int j = 0; j < 6; j++) { v[j] = r[tid + j * 256]; ss += v[j] * v[j]; }
#pragma unroll
  for (int off = 16; off; off >>= 1) ss += __shfl_xor(ss, off, 32);
  if ((tid & 31) == 0) red[tid >> 5] = ss;
  __syncthreads();
  float tot = 0.f;
#pragma unroll
  for (int i = 0; i < 8; i++) tot += red[i];
  float rr = w ? rsqrtf(tot * (1.0f / DMODEL) + RMS_EPS) : 1.0f;
#pragma unroll
  for (int j = 0; j < 6; j++) {
    int d = tid + j * 256;
    vbuf[d] = w ? (v[j] * rr * w[d]) : v[j];
  }
  __syncthreads();
#pragma unroll
  for (int j = 0; j < 3; j++) {           // 768 (even,odd) pairs per row
    int p = tid + j * 256;
    int h = p >> 6, i = p & 63;
    float a = vbuf[h * HDIM + 2 * i];
    float b = vbuf[h * HDIM + 2 * i + 1];
    float re = a, im = b;
    if (freqs) {
      float fr = freqs[(size_t)row * (HDIM / 2) + i];
      float c = __cosf(fr), s = __sinf(fr);
      re = a * c - b * s;
      im = a * s + b * c;
    }
    if (tr) {
      out[((size_t)h * HDIM + 2 * i)     * M + row] = f2bf(re);
      out[((size_t)h * HDIM + 2 * i + 1) * M + row] = f2bf(im);
    } else {
      bf16* o = out + ((size_t)h * M + row) * HDIM;
      o[2 * i]     = f2bf(re);
      o[2 * i + 1] = f2bf(im);
    }
  }
}

// ---------------- f32 -> bf16 convert ----------------
__global__ __launch_bounds__(256) void convert_kernel(const float* __restrict__ in,
                                                      bf16* __restrict__ out, int n)
{
  int i = blockIdx.x * 256 + threadIdx.x;
  if (i < n) out[i] = f2bf(in[i]);
}

// ---------------- residual: out = x + y * gate ----------------
__global__ __launch_bounds__(256) void resid_kernel(const float* __restrict__ x,
                                                    const float* __restrict__ y,
                                                    const float* __restrict__ ev,
                                                    float* __restrict__ out, int n)
{
  int i = blockIdx.x * 256 + threadIdx.x;
  if (i >= n) return;
  float g = ev ? ev[i % DMODEL] : 1.0f;
  out[i] = x[i] + y[i] * g;
}

// ---------------- bf16 WMMA GEMM: C[M,N] = A[M,K](bf16) * BT[N,K](bf16) + bias ----------------
// block tile 128x128, BK=64, double-buffered LDS filled by GLOBAL_LOAD_ASYNC_TO_LDS_B128.
// Pipeline per K-tile: s_wait_asynccnt 0 -> barrier -> issue async for next tile (other buf)
// -> 16 v_wmma on current buf. One barrier per tile, zero VGPR staging.
__global__ __launch_bounds__(256, 2) void gemm_kernel(
    const bf16* __restrict__ A, const bf16* __restrict__ BT,
    const float* __restrict__ bias, float* __restrict__ Cf,
    bf16* __restrict__ Cb, int M, int N, int K, int act)
{
  __shared__ bf16 As[2][128][72];     // [buf][m][k], padded (72*2B = 36 banks: conflict-free)
  __shared__ bf16 Bs[2][128][72];     // [buf][n][k]
  int tid = threadIdx.x;
  int bm = blockIdx.y * 128, bn = blockIdx.x * 128;
  int wv = tid >> 5, lane = tid & 31;
  int m0 = (wv >> 1) * 32, n0 = (wv & 1) * 64;
  const f32x8 zero8 = {0.f,0.f,0.f,0.f,0.f,0.f,0.f,0.f};
  f32x8 acc[2][4];
#pragma unroll
  for (int i = 0; i < 2; i++)
#pragma unroll
    for (int j = 0; j < 4; j++) acc[i][j] = zero8;

  // per-thread tile coords: 128x64 bf16 = 1024 16B chunks, 4 per thread
  int lm[4], lk[4];
#pragma unroll
  for (int i = 0; i < 4; i++) {
    int c = tid + i * 256;
    lm[i] = c >> 3;
    lk[i] = (c & 7) * 8;
  }

  auto issue = [&](int kt, int buf) {
#pragma unroll
    for (int i = 0; i < 4; i++) {
      unsigned la = (unsigned)(uintptr_t)&As[buf][lm[i]][lk[i]];
      unsigned lb = (unsigned)(uintptr_t)&Bs[buf][lm[i]][lk[i]];
      unsigned oa = (unsigned)(((size_t)(bm + lm[i]) * K + kt + lk[i]) * 2);
      unsigned ob = (unsigned)(((size_t)(bn + lm[i]) * K + kt + lk[i]) * 2);
      async_b128(la, oa, A);
      async_b128(lb, ob, BT);
    }
  };

  issue(0, 0);
  int nk = K / 64;
  for (int t = 0; t < nk; t++) {
    int buf = t & 1;
    wait_async0();                      // this wave's async fill of buf complete
    __syncthreads();                    // everyone's fill visible; prior readers of buf^1 done
    if (t + 1 < nk) issue((t + 1) * 64, buf ^ 1);
#pragma unroll
    for (int ks = 0; ks < 2; ks++) {
      bf16x16 af[2], bfr[4];
#pragma unroll
      for (int i = 0; i < 2; i++) af[i]  = load_frag(&As[buf][0][0], m0 + i * 16, ks * 32, 72);
#pragma unroll
      for (int j = 0; j < 4; j++) bfr[j] = load_frag(&Bs[buf][0][0], n0 + j * 16, ks * 32, 72);
#pragma unroll
      for (int i = 0; i < 2; i++)
#pragma unroll
        for (int j = 0; j < 4; j++) acc[i][j] = wmma_bf16(af[i], bfr[j], acc[i][j]);
    }
  }

  int colk = lane & 15, rhalf = (lane >> 4) << 3;
#pragma unroll
  for (int i = 0; i < 2; i++)
#pragma unroll
    for (int j = 0; j < 4; j++) {
      int gn = bn + n0 + j * 16 + colk;
      float bv = bias[gn];
#pragma unroll
      for (int rj = 0; rj < 8; rj++) {
        int gm = bm + m0 + i * 16 + rhalf + rj;
        float v = acc[i][j][rj] + bv;
        if (act) v = v * sigmoid_fast(1.5957691216057308f * (v + 0.044715f * v * v * v));
        size_t idx = (size_t)gm * N + gn;
        if (Cf) Cf[idx] = v;
        if (Cb) Cb[idx] = f2bf(v);
      }
    }
}

// ---------------- flash attention ----------------
// Q,K: [H][rows][HD] bf16; V: transposed [H][HD][Skv] bf16 -> Y[S][D] bf16
__global__ __launch_bounds__(256, 2) void attn_kernel(
    const bf16* __restrict__ Q, const bf16* __restrict__ Kh,
    const bf16* __restrict__ VT, const int* __restrict__ lens,
    bf16* __restrict__ Y, int Skv)
{
  __shared__ bf16 Qs[128][136];
  __shared__ bf16 Ks[32][136];
  __shared__ bf16 VTs[128][40];        // [hd][key]
  __shared__ bf16 Ps[8][16][40];       // per-wave P tile
  int tid = threadIdx.x, wv = tid >> 5, lane = tid & 31;
  int h = blockIdx.y;
  int qbase = blockIdx.x * 128;
  int len = lens[0];
  int colk = lane & 15, rhalf = (lane >> 4) << 3;

  const bf16* qp = Q + ((size_t)h * S_LEN + qbase) * HDIM;
#pragma unroll
  for (int i = 0; i < 8; i++) {                         // Q tile 128x128
    int c = tid + i * 256;
    int m = c >> 4, k0 = (c & 15) * 8;
    *(uint4*)(&Qs[m][k0]) = *(const uint4*)(qp + (size_t)m * HDIM + k0);
  }
  __syncthreads();
  bf16x16 qf[4];
#pragma unroll
  for (int kk = 0; kk < 4; kk++) qf[kk] = load_frag(&Qs[0][0], wv * 16, kk * 32, 136);

  const f32x8 zero8 = {0.f,0.f,0.f,0.f,0.f,0.f,0.f,0.f};
  f32x8 o[8];
  float mrow[8], lrow[8];
#pragma unroll
  for (int t = 0; t < 8; t++) o[t] = zero8;
#pragma unroll
  for (int j = 0; j < 8; j++) { mrow[j] = -1e30f; lrow[j] = 0.f; }

  const bf16* kpBase = Kh + (size_t)h * Skv * HDIM;
  const bf16* vpBase = VT + (size_t)h * HDIM * Skv;
  int ntiles = Skv / 32;

  // per-thread load coords: K tile 32x128 (m=key,k=hd), V tile 128x32 ([hd][key])
  int km[2], kk0[2], vh_[2], vk0[2];
#pragma unroll
  for (int i = 0; i < 2; i++) {
    int c = tid + i * 256;
    km[i]  = c >> 4;  kk0[i] = (c & 15) * 8;
    vh_[i] = c >> 2;  vk0[i] = (c & 3) * 8;
  }
  uint4 rk[2], rv[2];
#pragma unroll
  for (int i = 0; i < 2; i++) {                         // prefetch tile 0
    rk[i] = *(const uint4*)(kpBase + (size_t)km[i] * HDIM + kk0[i]);
    rv[i] = *(const uint4*)(vpBase + (size_t)vh_[i] * Skv + vk0[i]);
  }

  for (int kt = 0; kt < ntiles; kt++) {
#pragma unroll
    for (int i = 0; i < 2; i++) {
      *(uint4*)(&Ks[km[i]][kk0[i]])   = rk[i];
      *(uint4*)(&VTs[vh_[i]][vk0[i]]) = rv[i];
    }
    __syncthreads();
    if (kt + 1 < ntiles) {                              // prefetch next K/V tile
      int kb = (kt + 1) * 32;
#pragma unroll
      for (int i = 0; i < 2; i++) {
        rk[i] = *(const uint4*)(kpBase + (size_t)(kb + km[i]) * HDIM + kk0[i]);
        rv[i] = *(const uint4*)(vpBase + (size_t)vh_[i] * Skv + kb + vk0[i]);
      }
    }

    // S = Q K^T  (16 q-rows x 32 keys per wave)
    f32x8 s0 = zero8, s1 = zero8;
#pragma unroll
    for (int kk = 0; kk < 4; kk++) {
      bf16x16 b0 = load_frag(&Ks[0][0], 0,  kk * 32, 136);
      bf16x16 b1 = load_frag(&Ks[0][0], 16, kk * 32, 136);
      s0 = wmma_bf16(qf[kk], b0, s0);
      s1 = wmma_bf16(qf[kk], b1, s1);
    }

    // online softmax update
    int kg0 = kt * 32 + colk, kg1 = kg0 + 16;
    float corr[8];
#pragma unroll
    for (int j = 0; j < 8; j++) {
      float a0 = (kg0 < len) ? s0[j] * ATTN_SCALE : -1e30f;
      float a1 = (kg1 < len) ? s1[j] * ATTN_SCALE : -1e30f;
      float mt = fmaxf(a0, a1);
#pragma unroll
      for (int off = 8; off; off >>= 1) mt = fmaxf(mt, __shfl_xor(mt, off, 32));
      float nm = fmaxf(mrow[j], mt);
      float c  = __expf(mrow[j] - nm);
      float p0 = __expf(a0 - nm);
      float p1 = __expf(a1 - nm);
      float rs = p0 + p1;
#pragma unroll
      for (int off = 8; off; off >>= 1) rs += __shfl_xor(rs, off, 32);
      lrow[j] = lrow[j] * c + rs;
      mrow[j] = nm;
      corr[j] = c;
      Ps[wv][rhalf + j][colk]      = f2bf(p0);
      Ps[wv][rhalf + j][16 + colk] = f2bf(p1);
    }
    asm volatile("s_wait_dscnt 0x0" ::: "memory");      // P store -> P load (same wave, LDS in-order)

#pragma unroll
    for (int t = 0; t < 8; t++)
#pragma unroll
      for (int j = 0; j < 8; j++) o[t][j] *= corr[j];

    bf16x16 pf = load_frag(&Ps[wv][0][0], 0, 0, 40);
#pragma unroll
    for (int t = 0; t < 8; t++) {
      bf16x16 vf = load_frag(&VTs[0][0], t * 16, 0, 40);
      o[t] = wmma_bf16(pf, vf, o[t]);
    }
    __syncthreads();
  }

#pragma unroll
  for (int j = 0; j < 8; j++) lrow[j] = 1.0f / lrow[j];
  int srow = qbase + wv * 16 + rhalf;
#pragma unroll
  for (int t = 0; t < 8; t++) {
    int d = h * HDIM + t * 16 + colk;
#pragma unroll
    for (int j = 0; j < 8; j++)
      Y[(size_t)(srow + j) * DMODEL + d] = f2bf(o[t][j] * lrow[j]);
  }
}

// ---------------- host orchestration ----------------

extern "C" void kernel_launch(void* const* d_in, const int* in_sizes, int n_in,
                              void* d_out, int out_size, void* d_ws, size_t ws_size,
                              hipStream_t stream)
{
  (void)in_sizes; (void)n_in; (void)out_size; (void)ws_size;
  const float* x       = (const float*)d_in[0];
  const float* ctx     = (const float*)d_in[1];
  const float* e       = (const float*)d_in[2];   // (6,1,1,1,D)
  const float* freqs   = (const float*)d_in[3];
  const int*   seq_lens     = (const int*)d_in[4];
  const int*   context_lens = (const int*)d_in[5];
  const float* norm1_w = (const float*)d_in[7];
  const float* norm2_w = (const float*)d_in[8];
  const float* norm3_w = (const float*)d_in[9];
  const float* sa_qw = (const float*)d_in[10];
  const float* sa_qb = (const float*)d_in[11];
  const float* sa_kw = (const float*)d_in[12];
  const float* sa_kb = (const float*)d_in[13];
  const float* sa_vw = (const float*)d_in[14];
  const float* sa_vb = (const float*)d_in[15];
  const float* sa_nqw = (const float*)d_in[16];
  const float* sa_nkw = (const float*)d_in[17];
  const float* sa_ow = (const float*)d_in[18];
  const float* sa_ob = (const float*)d_in[19];
  const float* ca_qw = (const float*)d_in[20];
  const float* ca_qb = (const float*)d_in[21];
  const float* ca_kw = (const float*)d_in[22];
  const float* ca_kb = (const float*)d_in[23];
  const float* ca_vw = (const float*)d_in[24];
  const float* ca_vb = (const float*)d_in[25];
  const float* ca_nqw = (const float*)d_in[26];
  const float* ca_nkw = (const float*)d_in[27];
  const float* ca_ow = (const float*)d_in[28];
  const float* ca_ob = (const float*)d_in[29];
  const float* ffn_w1 = (const float*)d_in[30];
  const float* ffn_b1 = (const float*)d_in[31];
  const float* ffn_w2 = (const float*)d_in[32];
  const float* ffn_b2 = (const float*)d_in[33];
  float* out = (float*)d_out;

  const size_t SD = (size_t)S_LEN * DMODEL;
  const size_t LD = (size_t)L_LEN * DMODEL;
  const size_t SF = (size_t)S_LEN * FFDIM;
  const size_t DD = (size_t)DMODEL * DMODEL;
  const size_t DF = (size_t)DMODEL * FFDIM;
  char* w = (char*)d_ws;
  bf16*  xin_bf = (bf16*)w;  w += SD * 2;
  float* q_f    = (float*)w; w += SD * 4;
  float* k_f    = (float*)w; w += SD * 4;
  float* v_f    = (float*)w; w += SD * 4;
  bf16*  qh     = (bf16*)w;  w += SD * 2;
  bf16*  kh     = (bf16*)w;  w += SD * 2;
  bf16*  vh     = (bf16*)w;  w += SD * 2;   // transposed [H][HD][Skv]
  bf16*  y_bf   = (bf16*)w;  w += SD * 2;
  float* x1     = (float*)w; w += SD * 4;
  float* x2     = (float*)w; w += SD * 4;
  bf16*  ctx_bf = (bf16*)w;  w += LD * 2;
  bf16*  ff_bf  = (bf16*)w;  w += SF * 2;
  bf16*  wt_saq = (bf16*)w;  w += DD * 2;
  bf16*  wt_sak = (bf16*)w;  w += DD * 2;
  bf16*  wt_sav = (bf16*)w;  w += DD * 2;
  bf16*  wt_sao = (bf16*)w;  w += DD * 2;
  bf16*  wt_caq = (bf16*)w;  w += DD * 2;
  bf16*  wt_cak = (bf16*)w;  w += DD * 2;
  bf16*  wt_cav = (bf16*)w;  w += DD * 2;
  bf16*  wt_cao = (bf16*)w;  w += DD * 2;
  bf16*  wt_ff1 = (bf16*)w;  w += DF * 2;
  bf16*  wt_ff2 = (bf16*)w;  w += DF * 2;

  dim3 blk(256);
  dim3 gSD(DMODEL / 128, S_LEN / 128);
  dim3 gLD(DMODEL / 128, L_LEN / 128);
  dim3 gTD(DMODEL / 64, DMODEL / 64);

  // ---- weight pre-transpose/convert (read f32 once; GEMMs re-read bf16 from L2) ----
  transpose_conv_kernel<<<gTD, blk, 0, stream>>>(sa_qw, wt_saq, DMODEL, DMODEL);
  transpose_conv_kernel<<<gTD, blk, 0, stream>>>(sa_kw, wt_sak, DMODEL, DMODEL);
  transpose_conv_kernel<<<gTD, blk, 0, stream>>>(sa_vw, wt_sav, DMODEL, DMODEL);
  transpose_conv_kernel<<<gTD, blk, 0, stream>>>(sa_ow, wt_sao, DMODEL, DMODEL);
  transpose_conv_kernel<<<gTD, blk, 0, stream>>>(ca_qw, wt_caq, DMODEL, DMODEL);
  transpose_conv_kernel<<<gTD, blk, 0, stream>>>(ca_kw, wt_cak, DMODEL, DMODEL);
  transpose_conv_kernel<<<gTD, blk, 0, stream>>>(ca_vw, wt_cav, DMODEL, DMODEL);
  transpose_conv_kernel<<<gTD, blk, 0, stream>>>(ca_ow, wt_cao, DMODEL, DMODEL);
  transpose_conv_kernel<<<dim3(FFDIM / 64, DMODEL / 64), blk, 0, stream>>>(ffn_w1, wt_ff1, DMODEL, FFDIM);
  transpose_conv_kernel<<<dim3(DMODEL / 64, FFDIM / 64), blk, 0, stream>>>(ffn_w2, wt_ff2, FFDIM, DMODEL);

  // ---- self attention ----
  rmsmod_kernel<<<dim3(S_LEN), blk, 0, stream>>>(x, norm1_w, e + 0 * DMODEL, e + 1 * DMODEL, xin_bf);
  gemm_kernel<<<gSD, blk, 0, stream>>>(xin_bf, wt_saq, sa_qb, q_f, nullptr, S_LEN, DMODEL, DMODEL, 0);
  headprep_kernel<<<dim3(S_LEN), blk, 0, stream>>>(q_f, sa_nqw, freqs, qh, S_LEN, 0);
  gemm_kernel<<<gSD, blk, 0, stream>>>(xin_bf, wt_sak, sa_kb, k_f, nullptr, S_LEN, DMODEL, DMODEL, 0);
  headprep_kernel<<<dim3(S_LEN), blk, 0, stream>>>(k_f, sa_nkw, freqs, kh, S_LEN, 0);
  gemm_kernel<<<gSD, blk, 0, stream>>>(xin_bf, wt_sav, sa_vb, v_f, nullptr, S_LEN, DMODEL, DMODEL, 0);
  headprep_kernel<<<dim3(S_LEN), blk, 0, stream>>>(v_f, nullptr, nullptr, vh, S_LEN, 1);
  attn_kernel<<<dim3(S_LEN / 128, NHEAD), blk, 0, stream>>>(qh, kh, vh, seq_lens, y_bf, S_LEN);
  gemm_kernel<<<gSD, blk, 0, stream>>>(y_bf, wt_sao, sa_ob, q_f, nullptr, S_LEN, DMODEL, DMODEL, 0);
  resid_kernel<<<dim3((int)(SD / 256)), blk, 0, stream>>>(x, q_f, e + 2 * DMODEL, x1, (int)SD);

  // ---- cross attention ----
  rmsmod_kernel<<<dim3(S_LEN), blk, 0, stream>>>(x1, norm3_w, nullptr, nullptr, xin_bf);
  gemm_kernel<<<gSD, blk, 0, stream>>>(xin_bf, wt_caq, ca_qb, q_f, nullptr, S_LEN, DMODEL, DMODEL, 0);
  headprep_kernel<<<dim3(S_LEN), blk, 0, stream>>>(q_f, ca_nqw, nullptr, qh, S_LEN, 0);
  convert_kernel<<<dim3((int)(LD / 256)), blk, 0, stream>>>(ctx, ctx_bf, (int)LD);
  gemm_kernel<<<gLD, blk, 0, stream>>>(ctx_bf, wt_cak, ca_kb, k_f, nullptr, L_LEN, DMODEL, DMODEL, 0);
  headprep_kernel<<<dim3(L_LEN), blk, 0, stream>>>(k_f, ca_nkw, nullptr, kh, L_LEN, 0);
  gemm_kernel<<<gLD, blk, 0, stream>>>(ctx_bf, wt_cav, ca_vb, v_f, nullptr, L_LEN, DMODEL, DMODEL, 0);
  headprep_kernel<<<dim3(L_LEN), blk, 0, stream>>>(v_f, nullptr, nullptr, vh, L_LEN, 1);
  attn_kernel<<<dim3(S_LEN / 128, NHEAD), blk, 0, stream>>>(qh, kh, vh, context_lens, y_bf, L_LEN);
  gemm_kernel<<<gSD, blk, 0, stream>>>(y_bf, wt_cao, ca_ob, q_f, nullptr, S_LEN, DMODEL, DMODEL, 0);
  resid_kernel<<<dim3((int)(SD / 256)), blk, 0, stream>>>(x1, q_f, nullptr, x2, (int)SD);

  // ---- FFN ----
  rmsmod_kernel<<<dim3(S_LEN), blk, 0, stream>>>(x2, norm2_w, e + 3 * DMODEL, e + 4 * DMODEL, xin_bf);
  gemm_kernel<<<dim3(FFDIM / 128, S_LEN / 128), blk, 0, stream>>>(xin_bf, wt_ff1, ffn_b1, nullptr, ff_bf,
                                                                 S_LEN, FFDIM, DMODEL, 1);
  gemm_kernel<<<gSD, blk, 0, stream>>>(ff_bf, wt_ff2, ffn_b2, q_f, nullptr, S_LEN, DMODEL, FFDIM, 0);
  resid_kernel<<<dim3((int)(SD / 256)), blk, 0, stream>>>(x2, q_f, e + 5 * DMODEL, out, (int)SD);
}
